// Detector_52518860095983
// MI455X (gfx1250) — matrix operand order repaired and checked by reading, not compiled, not run on hardware
//
#include <hip/hip_runtime.h>
#include <stdint.h>

typedef _Float16 v16h __attribute__((ext_vector_type(16)));
typedef float    v8f  __attribute__((ext_vector_type(8)));

#define FIND_T 0.5f
#define IOU_T  0.5f
#define DET_T  0.5f

constexpr int KTOP = 512;
constexpr int C13  = 32 * 3 * 13 * 13;   // 16224
constexpr int C26  = 32 * 3 * 26 * 26;   // 64896
constexpr int C52  = 32 * 3 * 52 * 52;   // 259584
constexpr int NTOT = C13 + C26 + C52;    // 340704

struct SelState {
  unsigned hist[256];
  unsigned long long prefix;
  unsigned k_rem;
  unsigned cnt;
  unsigned long long kth;
};

// workspace layout
constexpr size_t OFF_KEYS  = 0;
constexpr size_t OFF_BOXES = (size_t)NTOT * 8;                 // keys: u64[NTOT]
constexpr size_t OFF_STATE = OFF_BOXES + (size_t)NTOT * 24;    // boxes: f32[NTOT*6]
constexpr size_t OFF_TOP   = OFF_STATE + 2048;                 // SelState (padded)

__device__ __forceinline__ unsigned f2ord(float f) {
  unsigned b = __float_as_uint(f);
  return (b & 0x80000000u) ? ~b : (b | 0x80000000u);
}
__device__ __forceinline__ float ord2f(unsigned u) {
  unsigned b = (u & 0x80000000u) ? (u & 0x7FFFFFFFu) : ~u;
  return __uint_as_float(b);
}

// ---------------- decode: heads -> (boxes, sort keys) --------------------
// CDNA5 path: class logits streamed to LDS with double-buffered async loads.
__global__ __launch_bounds__(256) void k_decode(
    const float* __restrict__ o13, const float* __restrict__ o26,
    const float* __restrict__ o52, const float* __restrict__ a13,
    const float* __restrict__ a26, const float* __restrict__ a52,
    const float* __restrict__ c1p, const float* __restrict__ c2p,
    unsigned long long* __restrict__ keys, float* __restrict__ boxes) {
  __shared__ float cbuf[2][16][256];
  const int tid = threadIdx.x;
  const int t   = blockIdx.x * 256 + tid;
  const int tt  = (t < NTOT) ? t : (NTOT - 1);

  const float* src; const float* anc; int S, local, candBase;
  if (tt < C13)            { S = 13; src = o13; anc = a13; local = tt;             candBase = 0;         }
  else if (tt < C13 + C26) { S = 26; src = o26; anc = a26; local = tt - C13;       candBase = C13;       }
  else                     { S = 52; src = o52; anc = a52; local = tt - C13 - C26; candBase = C13 + C26; }
  const int S2 = S * S;
  const int b   = local / (3 * S2);
  const int r   = local - b * 3 * S2;
  const int a   = r / S2;
  const int pos = r - a * S2;

  const float* p0 = src + (size_t)(b * 255 + a * 85) * S2 + pos;
  const float o0 = p0[0];
  const float o1 = p0[(size_t)S2];
  const float o2 = p0[(size_t)2 * S2];
  const float o3 = p0[(size_t)3 * S2];
  const float o4 = p0[(size_t)4 * S2];
  const float* cls0 = p0 + (size_t)5 * S2;

  auto issue = [&](int chunk, int buf) {
#pragma unroll
    for (int f = 0; f < 16; ++f) {
      const float* g = cls0 + (size_t)(chunk * 16 + f) * S2;
      unsigned loff = (unsigned)(uintptr_t)&cbuf[buf][f][tid];  // low 32b of generic ptr == LDS offset
      asm volatile("global_load_async_to_lds_b32 %0, %1, off"
                   :: "v"(loff), "v"(g) : "memory");
    }
  };

  issue(0, 0);
  float m = -1e30f, ssum = 0.0f, best = -1e30f;
  int bi = 0;
  for (int c = 0; c < 5; ++c) {                 // 5 chunks x 16 = 80 classes
    const int buf = c & 1;
    if (c < 4) {
      issue(c + 1, buf ^ 1);
      asm volatile("s_wait_asynccnt 16" ::: "memory");   // chunk c complete
    } else {
      asm volatile("s_wait_asynccnt 0" ::: "memory");
    }
#pragma unroll
    for (int f = 0; f < 16; ++f) {
      const float v = cbuf[buf][f][tid];
      const int ci = c * 16 + f;
      if (v > best) { best = v; bi = ci; }
      if (v > m) { ssum = ssum * __expf(m - v) + 1.0f; m = v; }
      else       { ssum += __expf(v - m); }
    }
  }

  const float case1 = *c1p, case2 = *c2p;
  const float obj = 1.0f / (1.0f + __expf(-o0));
  const float p   = obj / ssum;                 // obj * max(softmax)
  const float s1  = 416.0f / (float)S;          // DATA_W/S == DATA_H/S
  const int   x = pos % S, y = pos / S;
  const float cx = ((float)x + o1) * s1 / case1;
  const float cy = ((float)y + o2) * s1 / case2;
  const float w  = __expf(o3) * anc[a * 2 + 0] / case1;
  const float h  = __expf(o4) * anc[a * 2 + 1] / case2;
  const float sc = (obj > FIND_T) ? p : -1.0f;

  if (t < NTOT) {
    const int n = candBase + (b * S2 + pos) * 3 + a;   // reference concat order
    float* bp = boxes + (size_t)n * 6;
    bp[0] = p; bp[1] = cx; bp[2] = cy; bp[3] = w; bp[4] = h; bp[5] = (float)bi;
    keys[n] = ((unsigned long long)f2ord(sc) << 32) |
              (unsigned long long)(0xFFFFFFFFu - (unsigned)n);
  }
}

// ---------------- radix select (exact top-512 of distinct 64-bit keys) ----
__global__ void k_init(SelState* st) {
  int t = threadIdx.x;
  st->hist[t] = 0;
  if (t == 0) { st->prefix = 0ull; st->k_rem = KTOP; st->cnt = 0; st->kth = 0ull; }
}

__global__ __launch_bounds__(256) void k_hist(const unsigned long long* __restrict__ keys,
                                              SelState* st, int shift, int pass) {
  int i = blockIdx.x * 256 + threadIdx.x;
  if (i >= NTOT) return;
  unsigned long long k = keys[i];
  if (pass == 0 || (k >> (shift + 8)) == (st->prefix >> (shift + 8)))
    atomicAdd(&st->hist[(unsigned)((k >> shift) & 0xFFull)], 1u);
}

__global__ void k_scan(SelState* st, int shift, int last) {
  __shared__ unsigned h[256];
  int t = threadIdx.x;
  h[t] = st->hist[t];
  __syncthreads();
  if (t == 0) {
    unsigned rem = st->k_rem;
    unsigned above = 0;
    int chosen = 0;
    for (int bb = 255; bb >= 0; --bb) {
      unsigned c2 = above + h[bb];
      if (c2 >= rem) { chosen = bb; break; }
      above = c2;
    }
    unsigned long long np = st->prefix | ((unsigned long long)chosen << shift);
    st->prefix = np;
    st->k_rem = rem - above;
    if (last) { st->kth = np; st->cnt = 0; }
  }
  st->hist[t] = 0;  // ready for next pass
}

__global__ __launch_bounds__(256) void k_collect(const unsigned long long* __restrict__ keys,
                                                 SelState* st,
                                                 unsigned long long* __restrict__ topk) {
  int i = blockIdx.x * 256 + threadIdx.x;
  if (i >= NTOT) return;
  unsigned long long k = keys[i];
  if (k >= st->kth) {
    unsigned pos = atomicAdd(&st->cnt, 1u);
    if (pos < KTOP) topk[pos] = k;   // exactly 512 distinct keys >= kth
  }
}

// ---------------- final: sort, filter, IoU + WMMA same-class, NMS --------
__global__ __launch_bounds__(512) void k_final(const float* __restrict__ boxes,
                                               const unsigned long long* __restrict__ topk,
                                               float* __restrict__ out) {
  __shared__ unsigned long long sk[512];
  __shared__ float s_p[512], s_cx[512], s_cy[512], s_w[512], s_h[512];
  __shared__ int   s_cls[512];
  __shared__ float s_x1[512], s_y1[512], s_x2[512], s_y2[512];
  __shared__ unsigned s_sup[512][16];   // 512x512 suppression bitmask
  __shared__ float s_red[512];          // max-reduce, then keep flags
  __shared__ unsigned s_suppr[512];
  __shared__ unsigned s_act;
  const int tid = threadIdx.x;

  sk[tid] = topk[tid];
  // bitonic sort, descending (u64 key = score-ordered | inverted index)
  for (unsigned k = 2; k <= 512; k <<= 1)
    for (unsigned j = k >> 1; j > 0; j >>= 1) {
      __syncthreads();
      unsigned ixj = (unsigned)tid ^ j;
      if (ixj > (unsigned)tid) {
        unsigned long long va = sk[tid], vb = sk[ixj];
        bool sw = (((unsigned)tid & k) == 0) ? (va < vb) : (va > vb);
        if (sw) { sk[tid] = vb; sk[ixj] = va; }
      }
    }
  __syncthreads();

  const unsigned long long key = sk[tid];
  const unsigned n = 0xFFFFFFFFu - (unsigned)(key & 0xFFFFFFFFull);
  const float sc = ord2f((unsigned)(key >> 32));
  const float* bp = boxes + (size_t)n * 6;
  const float pv = bp[0], cx = bp[1], cy = bp[2], w = bp[3], h = bp[4], clf = bp[5];
  s_p[tid] = pv; s_cx[tid] = cx; s_cy[tid] = cy; s_w[tid] = w; s_h[tid] = h;
  s_cls[tid] = (int)clf;
  const float x1 = cx - w * 0.5f, y1 = cy - h * 0.5f;
  s_x1[tid] = x1; s_y1[tid] = y1; s_x2[tid] = x1 + w; s_y2[tid] = y1 + h;
  const bool alive0 = sc > 0.0f;
  s_red[tid] = alive0 ? pv : 0.0f;
  __syncthreads();
  for (int s = 256; s > 0; s >>= 1) {
    if (tid < s) s_red[tid] = fmaxf(s_red[tid], s_red[tid + s]);
    __syncthreads();
  }
  const float maxp = s_red[0];
  __syncthreads();
  const bool alive = alive0 && (pv > DET_T * maxp);
  s_suppr[tid] = alive ? 0u : 1u;

  // IoU >= 0.5 bits (thread owns its row)
  const float rx1 = x1, ry1 = y1, rx2 = x1 + w, ry2 = y1 + h, rar = w * h;
  for (int cw = 0; cw < 16; ++cw) {
    unsigned bits = 0;
    for (int bb = 0; bb < 32; ++bb) {
      const int j = cw * 32 + bb;
      float iw = fminf(rx2, s_x2[j]) - fmaxf(rx1, s_x1[j]); iw = fmaxf(iw, 0.0f);
      float ih = fminf(ry2, s_y2[j]) - fmaxf(ry1, s_y1[j]); ih = fmaxf(ih, 0.0f);
      const float inter = iw * ih;
      const float iou = inter / (rar + s_w[j] * s_h[j] - inter);
      if (iou >= IOU_T) bits |= (1u << bb);
    }
    s_sup[tid][cw] = bits;
  }
  __syncthreads();

  // same_cls = OneHot(cls) @ OneHot(cls)^T  via v_wmma_f32_16x16x32_f16
  // 80 classes padded to K=96 (3 chunks of 32). Exact 0/1 arithmetic.
  // Wave owns 2 tile-rows (tr = wv, wv+16): A fragments built once per row,
  // reused across all 32 tile-columns.
  const int lane = tid & 31, wv = tid >> 5;
  for (int tr = wv; tr < 32; tr += 16) {
    const int clsR = s_cls[tr * 16 + (lane & 15)];
    v16h Afrag[3];
#pragma unroll
    for (int ch = 0; ch < 3; ++ch) {
      const int kb = ch * 32;
#pragma unroll
      for (int hh = 0; hh < 16; ++hh) {
        // A 16x32 f16 layout: lanes 0-15 -> K {0..7,16..23}; lanes 16-31 -> K {8..15,24..31}
        const int K = kb + ((lane < 16) ? ((hh < 8) ? hh : hh + 8)
                                        : ((hh < 8) ? hh + 8 : hh + 16));
        Afrag[ch][hh] = (clsR == K) ? (_Float16)1.0f : (_Float16)0.0f;
      }
    }
    const int row0 = tr * 16 + ((lane >> 4) << 3);   // D: M = r + 8*(lane>=16)
    for (int tc = 0; tc < 32; ++tc) {
      v8f acc = {0.f, 0.f, 0.f, 0.f, 0.f, 0.f, 0.f, 0.f};
#pragma unroll
      for (int ch = 0; ch < 3; ++ch) {
        v16h Bm;
        const int kbl = ch * 32;                     // B 32x16 f16: N = lane%16, K as A
#pragma unroll
        for (int hh = 0; hh < 16; ++hh)              // half index = K (same order as A)
          Bm[hh] = (s_cls[tc * 16 + (lane & 15)] == kbl + ((lane < 16) ? ((hh < 8) ? hh : hh + 8) : ((hh < 8) ? hh + 8 : hh + 16))) ? (_Float16)1.0f : (_Float16)0.0f;
        acc = __builtin_amdgcn_wmma_f32_16x16x32_f16(false, Afrag[ch], false, Bm,
                                                     (short)0, acc, false, false);
      }
      const int col = tc * 16 + (lane & 15);         // D: N = lane%16
#pragma unroll
      for (int rr = 0; rr < 8; ++rr)
        if (acc[rr] > 0.5f)
          atomicOr(&s_sup[row0 + rr][col >> 5], 1u << (col & 31));
    }
  }
  __syncthreads();

  // sequential NMS exactly as reference fori_loop
  for (int i = 0; i < 512; ++i) {
    if (tid == i) {
      unsigned act = s_suppr[i] ? 0u : 1u;
      s_red[i] = (float)act;     // keep flag
      s_act = act;
    }
    __syncthreads();
    if (s_act && tid > i && ((s_sup[i][tid >> 5] >> (tid & 31)) & 1u))
      s_suppr[tid] = 1u;
    __syncthreads();
  }

  const float kf = s_red[tid];
  out[tid * 6 + 0] = pv * kf;
  out[tid * 6 + 1] = cx * kf;
  out[tid * 6 + 2] = cy * kf;
  out[tid * 6 + 3] = w * kf;
  out[tid * 6 + 4] = h * kf;
  out[tid * 6 + 5] = clf * kf;
  out[3072 + tid]  = kf;
}

extern "C" void kernel_launch(void* const* d_in, const int* in_sizes, int n_in,
                              void* d_out, int out_size, void* d_ws, size_t ws_size,
                              hipStream_t stream) {
  (void)in_sizes; (void)n_in; (void)out_size; (void)ws_size;
  const float* o13 = (const float*)d_in[0];
  const float* o26 = (const float*)d_in[1];
  const float* o52 = (const float*)d_in[2];
  const float* a13 = (const float*)d_in[3];
  const float* a26 = (const float*)d_in[4];
  const float* a52 = (const float*)d_in[5];
  const float* c1  = (const float*)d_in[6];
  const float* c2  = (const float*)d_in[7];

  char* ws = (char*)d_ws;
  unsigned long long* keys = (unsigned long long*)(ws + OFF_KEYS);
  float* boxes             = (float*)(ws + OFF_BOXES);
  SelState* st             = (SelState*)(ws + OFF_STATE);
  unsigned long long* topk = (unsigned long long*)(ws + OFF_TOP);
  float* out               = (float*)d_out;

  const int blocks = (NTOT + 255) / 256;  // 1331

  k_init<<<1, 256, 0, stream>>>(st);
  k_decode<<<blocks, 256, 0, stream>>>(o13, o26, o52, a13, a26, a52, c1, c2,
                                       keys, boxes);
  for (int p = 0; p < 8; ++p) {
    const int shift = 56 - 8 * p;
    k_hist<<<blocks, 256, 0, stream>>>(keys, st, shift, p);
    k_scan<<<1, 256, 0, stream>>>(st, shift, (p == 7) ? 1 : 0);
  }
  k_collect<<<blocks, 256, 0, stream>>>(keys, st, topk);
  k_final<<<1, 512, 0, stream>>>(boxes, topk, out);
}
